// CascadeEmbedding_50199577755789
// MI455X (gfx1250) — compile-verified, matrix operand order checked by reading, not run on hardware
//
#include <hip/hip_runtime.h>
#include <hip/hip_bf16.h>

// ---------------------------------------------------------------------------
// Problem constants (from reference)
// ---------------------------------------------------------------------------
constexpr int cV = 50000, cP = 512, cNC = 1000;
constexpr int cH = 768, cE = 256, cNH = 12, cHD = 64, cFF = 3072, cNL = 3;
constexpr int cB = 4, cS = 512, cNN = 13;          // NNODES
constexpr int cRows = cB * cS;                     // 2048 token rows
constexpr int cKcas = cNN * cE;                    // 3328

// GEMM epilogue flags
constexpr int kRelu   = 1;
constexpr int kTransB = 2;
constexpr int kBias   = 4;

typedef __bf16 bf16;
typedef __attribute__((ext_vector_type(16))) __bf16 v16bf;
typedef __attribute__((ext_vector_type(8)))  float  v8f;
typedef __attribute__((ext_vector_type(4)))  int    i4v;

// ---------------------------------------------------------------------------
// CDNA5 async Global->LDS path (ASYNCcnt), guarded so the file compiles on
// toolchains that do not expose the builtins. Builtin signature (from clang
// diagnostic): (global int4*, local int4*, imm offset, imm cpol).
// ---------------------------------------------------------------------------
#if __has_builtin(__builtin_amdgcn_global_load_async_to_lds_b128)
#define HAS_ASYNC_LDS 1
typedef __attribute__((address_space(1))) i4v* gas_i4p;
typedef __attribute__((address_space(3))) i4v* las_i4p;
#define GPTR(p) ((gas_i4p)(p))
#define LPTR(p) ((las_i4p)(p))
#else
#define HAS_ASYNC_LDS 0
#endif

__device__ __forceinline__ void wait_async_zero() {
#if __has_builtin(__builtin_amdgcn_s_wait_asynccnt)
  __builtin_amdgcn_s_wait_asynccnt(0);
#else
  asm volatile("s_wait_asynccnt 0x0" ::: "memory");
#endif
}

// Build a 16x32 bf16 WMMA fragment slice for one lane from an f32 LDS row.
// Element e holds K = (e<8?0:16) + g*8 + (e&7), per the CDNA5 16-bit layout.
__device__ __forceinline__ v16bf make_frag(const float* row, int g) {
  const float4 x0 = *(const float4*)(row + g * 8);
  const float4 x1 = *(const float4*)(row + g * 8 + 4);
  const float4 y0 = *(const float4*)(row + 16 + g * 8);
  const float4 y1 = *(const float4*)(row + 16 + g * 8 + 4);
  v16bf f;
  f[0]  = (bf16)x0.x; f[1]  = (bf16)x0.y; f[2]  = (bf16)x0.z; f[3]  = (bf16)x0.w;
  f[4]  = (bf16)x1.x; f[5]  = (bf16)x1.y; f[6]  = (bf16)x1.z; f[7]  = (bf16)x1.w;
  f[8]  = (bf16)y0.x; f[9]  = (bf16)y0.y; f[10] = (bf16)y0.z; f[11] = (bf16)y0.w;
  f[12] = (bf16)y1.x; f[13] = (bf16)y1.y; f[14] = (bf16)y1.z; f[15] = (bf16)y1.w;
  return f;
}

// ---------------------------------------------------------------------------
// Generic tiled GEMM:  C[z][M,N] = alpha * A[z][M,K] @ B[z][N,K]^T (+bias)(relu)
//   - B given as [N,K] row-major (PyTorch weight layout); with kTransB,
//     B[n,k] is read from Bm[k*ldb + n] (used for attn @ V).
//   - batch z decomposed as zo=z/zdiv, zi=z%zdiv with separate strides so
//     attention (b,h) offsets work with one kernel.
//   - f32 tiles staged to LDS with GLOBAL_LOAD_ASYNC_TO_LDS_B128, converted
//     to bf16 on fragment build, v_wmma_f32_16x16x32_bf16 accumulation.
// Tile: 64x64x32, 128 threads (4 waves), each wave owns a 32x32 sub-tile.
// ---------------------------------------------------------------------------
__global__ __launch_bounds__(128) void gemm_bf16_wmma(
    const float* __restrict__ A, const float* __restrict__ Bm,
    const float* __restrict__ bias, float* __restrict__ C,
    int M, int N, int K, int lda, int ldb, int ldc,
    long long aOut, long long aIn, long long bOut, long long bIn,
    long long cOut, long long cIn, int zdiv, float alpha, int flags)
{
  __shared__ float Af[64][32];
  __shared__ float Bf[64][32];

  const int tid  = threadIdx.x;
  const int lane = tid & 31;
  const int wave = tid >> 5;
  const int wm   = (wave >> 1) * 32;   // wave's M offset in tile
  const int wn   = (wave & 1) * 32;    // wave's N offset in tile

  const int z  = blockIdx.z;
  const int zo = z / zdiv;
  const int zi = z - zo * zdiv;
  const float* Ab = A  + zo * aOut + zi * aIn;
  const float* Bb = Bm + zo * bOut + zi * bIn;
  float*       Cb = C  + zo * cOut + zi * cIn;

  const int m0 = blockIdx.y * 64;
  const int n0 = blockIdx.x * 64;

  v8f acc[2][2];
  #pragma unroll
  for (int i = 0; i < 2; ++i)
    #pragma unroll
    for (int j = 0; j < 2; ++j)
      acc[i][j] = v8f{0.f, 0.f, 0.f, 0.f, 0.f, 0.f, 0.f, 0.f};

  const int g    = lane >> 4;   // K-half selector (0/1)
  const int mrow = lane & 15;   // M (A) or N (B) index within 16

  // 64x32 f32 tile = 512 float4 transfers; 4 per thread, fixed trip count.
  const int r_st  = tid >> 3;          // rows this thread touches: r_st + 16*it
  const int c4_st = (tid & 7) * 4;

  for (int k0 = 0; k0 < K; k0 += 32) {
    // ---- stage A tile (64x32 f32) via async Global->LDS ----
    #pragma unroll
    for (int it = 0; it < 4; ++it) {
      const int r = r_st + it * 16;
      const float* ga = Ab + (long long)(m0 + r) * lda + k0 + c4_st;
#if HAS_ASYNC_LDS
      __builtin_amdgcn_global_load_async_to_lds_b128(GPTR(ga), LPTR(&Af[r][c4_st]), 0, 0);
#else
      *(float4*)&Af[r][c4_st] = *(const float4*)ga;
#endif
    }
    // ---- stage B tile (64x32 f32 as [N][K]) ----
    if (flags & kTransB) {
      // strided gather: B[n][k] = Bb[k*ldb + n]
      #pragma unroll
      for (int it = 0; it < 16; ++it) {
        const int t = it * 128 + tid;
        const int r = t >> 5, c = t & 31;
        Bf[r][c] = Bb[(long long)(k0 + c) * ldb + (n0 + r)];
      }
    } else {
      #pragma unroll
      for (int it = 0; it < 4; ++it) {
        const int r = r_st + it * 16;
        const float* gb = Bb + (long long)(n0 + r) * ldb + k0 + c4_st;
#if HAS_ASYNC_LDS
        __builtin_amdgcn_global_load_async_to_lds_b128(GPTR(gb), LPTR(&Bf[r][c4_st]), 0, 0);
#else
        *(float4*)&Bf[r][c4_st] = *(const float4*)gb;
#endif
      }
    }
    // hint next K tiles toward the caches (global_prefetch_b8)
    if (k0 + 32 < K) {
      __builtin_prefetch(Ab + (long long)(m0 + (tid >> 1)) * lda + k0 + 32 + (tid & 1) * 16, 0, 1);
      if (!(flags & kTransB))
        __builtin_prefetch(Bb + (long long)(n0 + (tid >> 1)) * ldb + k0 + 32 + (tid & 1) * 16, 0, 1);
    }
#if HAS_ASYNC_LDS
    wait_async_zero();
#endif
    __syncthreads();

    // ---- build bf16 fragments from f32 LDS rows, issue WMMA ----
    v16bf afr[2], bfr[2];
    #pragma unroll
    for (int i = 0; i < 2; ++i)
      afr[i] = make_frag(&Af[wm + i * 16 + mrow][0], g);
    #pragma unroll
    for (int j = 0; j < 2; ++j)
      bfr[j] = make_frag(&Bf[wn + j * 16 + mrow][0], g);

    #pragma unroll
    for (int i = 0; i < 2; ++i)
      #pragma unroll
      for (int j = 0; j < 2; ++j)
        acc[i][j] = __builtin_amdgcn_wmma_f32_16x16x32_bf16(
            false, afr[i], false, bfr[j], (short)0, acc[i][j], false, false);
    __syncthreads();
  }

  // ---- epilogue: lane -> N, VGPR r -> M = g*8 + r ----
  #pragma unroll
  for (int i = 0; i < 2; ++i) {
    #pragma unroll
    for (int j = 0; j < 2; ++j) {
      const int n = n0 + wn + j * 16 + mrow;
      const float bv = (flags & kBias) ? bias[n] : 0.f;
      #pragma unroll
      for (int r = 0; r < 8; ++r) {
        const int m = m0 + wm + i * 16 + g * 8 + r;
        float v = acc[i][j][r] * alpha + bv;
        if (flags & kRelu) v = fmaxf(v, 0.f);
        Cb[(long long)m * ldc + n] = v;
      }
    }
  }
}

// ---------------------------------------------------------------------------
// x[row, :] = tok_emb[ids[row], :] + pos_emb[row % S, :]
// ---------------------------------------------------------------------------
__global__ __launch_bounds__(256) void embed_kernel(
    const int* __restrict__ ids, const float* __restrict__ tok,
    const float* __restrict__ pos, float* __restrict__ x)
{
  const int row = blockIdx.x;
  const long long tb = (long long)ids[row] * cH;
  const long long pb = (long long)(row % cS) * cH;
  const long long ob = (long long)row * cH;
  #pragma unroll
  for (int e = 0; e < 3; ++e) {
    const int c = e * 256 + threadIdx.x;
    x[ob + c] = tok[tb + c] + pos[pb + c];
  }
}

// ---------------------------------------------------------------------------
// nemb[s,n,e] = mask ? node_emb[cid[s,n],e] + w[s,n]*cw_W[e] + cw_b[e] : 0
// ---------------------------------------------------------------------------
__global__ __launch_bounds__(256) void cascade_node_kernel(
    const int* __restrict__ cid, const float* __restrict__ w,
    const unsigned char* __restrict__ mask, const float* __restrict__ nodeEmb,
    const float* __restrict__ cwW, const float* __restrict__ cwB,
    float* __restrict__ out)
{
  const long long idx = (long long)blockIdx.x * 256 + threadIdx.x;
  if (idx >= (long long)cS * cNN * cE) return;
  const int e  = (int)(idx & (cE - 1));
  const int sn = (int)(idx >> 8);          // E == 256
  float v = 0.f;
  if (mask[sn]) {
    v = nodeEmb[(long long)cid[sn] * cE + e] + w[sn] * cwW[e] + cwB[e];
  }
  out[idx] = v;
}

// ---------------------------------------------------------------------------
// out[row,:] = LayerNorm(x[row,:] + delta[row % deltaMod, :]) * g + b
// ---------------------------------------------------------------------------
__global__ __launch_bounds__(256) void add_ln_kernel(
    const float* __restrict__ xin, const float* __restrict__ delta, int deltaMod,
    const float* __restrict__ g, const float* __restrict__ b, float eps,
    float* __restrict__ out)
{
  __shared__ float red[8];
  const int row = blockIdx.x;
  const int tid = threadIdx.x;
  const long long xo = (long long)row * cH;
  const long long dr = (long long)(row % deltaMod) * cH;

  float v[3];
  float s = 0.f;
  #pragma unroll
  for (int e = 0; e < 3; ++e) {
    const int c = e * 256 + tid;
    v[e] = xin[xo + c] + delta[dr + c];
    s += v[e];
  }
  #pragma unroll
  for (int o = 16; o > 0; o >>= 1) s += __shfl_xor(s, o, 32);
  if ((tid & 31) == 0) red[tid >> 5] = s;
  __syncthreads();
  if (tid == 0) { float t = 0.f; for (int w = 0; w < 8; ++w) t += red[w]; red[0] = t; }
  __syncthreads();
  const float mean = red[0] / (float)cH;
  __syncthreads();

  float s2 = 0.f;
  #pragma unroll
  for (int e = 0; e < 3; ++e) { const float d = v[e] - mean; s2 += d * d; }
  #pragma unroll
  for (int o = 16; o > 0; o >>= 1) s2 += __shfl_xor(s2, o, 32);
  if ((tid & 31) == 0) red[tid >> 5] = s2;
  __syncthreads();
  if (tid == 0) { float t = 0.f; for (int w = 0; w < 8; ++w) t += red[w]; red[0] = t; }
  __syncthreads();
  const float rstd = rsqrtf(red[0] / (float)cH + eps);

  #pragma unroll
  for (int e = 0; e < 3; ++e) {
    const int c = e * 256 + tid;
    out[xo + c] = (v[e] - mean) * rstd * g[c] + b[c];
  }
}

// ---------------------------------------------------------------------------
// In-place softmax over rows of length 512 (one block per row)
// ---------------------------------------------------------------------------
__global__ __launch_bounds__(256) void softmax_kernel(float* __restrict__ buf)
{
  __shared__ float red[8];
  float* p = buf + (long long)blockIdx.x * cS;
  const int tid = threadIdx.x;

  float v0 = p[tid], v1 = p[tid + 256];
  float m = fmaxf(v0, v1);
  #pragma unroll
  for (int o = 16; o > 0; o >>= 1) m = fmaxf(m, __shfl_xor(m, o, 32));
  if ((tid & 31) == 0) red[tid >> 5] = m;
  __syncthreads();
  if (tid == 0) { float t = red[0]; for (int w = 1; w < 8; ++w) t = fmaxf(t, red[w]); red[0] = t; }
  __syncthreads();
  m = red[0];
  __syncthreads();

  v0 = expf(v0 - m);
  v1 = expf(v1 - m);
  float s = v0 + v1;
  #pragma unroll
  for (int o = 16; o > 0; o >>= 1) s += __shfl_xor(s, o, 32);
  if ((tid & 31) == 0) red[tid >> 5] = s;
  __syncthreads();
  if (tid == 0) { float t = 0.f; for (int w = 0; w < 8; ++w) t += red[w]; red[0] = t; }
  __syncthreads();
  const float inv = 1.f / red[0];
  p[tid]       = v0 * inv;
  p[tid + 256] = v1 * inv;
}

// ---------------------------------------------------------------------------
// Workspace layout (f32 elements)
// ---------------------------------------------------------------------------
constexpr long long OFF_X    = 0;
constexpr long long SZ_X    = (long long)cRows * cH;               // 1.5M
constexpr long long OFF_NEMB = OFF_X + SZ_X;
constexpr long long SZ_NEMB = (long long)cS * cKcas;               // 1.7M
constexpr long long OFF_CASC = OFF_NEMB + SZ_NEMB;
constexpr long long SZ_CASC = (long long)cS * cH;
constexpr long long OFF_QKV  = OFF_CASC + SZ_CASC;
constexpr long long SZ_QKV  = (long long)cRows * 3 * cH;
constexpr long long OFF_SC   = OFF_QKV + SZ_QKV;
constexpr long long SZ_SC   = (long long)cB * cNH * cS * cS;       // 12.6M
constexpr long long OFF_CTX  = OFF_SC + SZ_SC;
constexpr long long OFF_PROJ = OFF_CTX + SZ_X;
constexpr long long OFF_FF   = OFF_PROJ + SZ_X;

extern "C" void kernel_launch(void* const* d_in, const int* in_sizes, int n_in,
                              void* d_out, int out_size, void* d_ws, size_t ws_size,
                              hipStream_t stream) {
  const int*   input_ids = (const int*)d_in[0];
  const int*   ccids     = (const int*)d_in[1];
  const float* cweights  = (const float*)d_in[2];
  const unsigned char* cmask = (const unsigned char*)d_in[3];
  const float* tok_emb   = (const float*)d_in[4];
  const float* pos_emb   = (const float*)d_in[5];
  const float* node_emb  = (const float*)d_in[6];
  const float* cw_W      = (const float*)d_in[7];
  const float* cw_b      = (const float*)d_in[8];
  const float* fus_W     = (const float*)d_in[9];
  const float* fus_b     = (const float*)d_in[10];
  const float* ln_g      = (const float*)d_in[11];
  const float* ln_b      = (const float*)d_in[12];
  const float* attn_in_W  = (const float*)d_in[13];
  const float* attn_in_b  = (const float*)d_in[14];
  const float* attn_out_W = (const float*)d_in[15];
  const float* attn_out_b = (const float*)d_in[16];
  const float* ff1_W = (const float*)d_in[17];
  const float* ff1_b = (const float*)d_in[18];
  const float* ff2_W = (const float*)d_in[19];
  const float* ff2_b = (const float*)d_in[20];
  const float* ln1_g = (const float*)d_in[21];
  const float* ln1_b = (const float*)d_in[22];
  const float* ln2_g = (const float*)d_in[23];
  const float* ln2_b = (const float*)d_in[24];

  float* ws    = (float*)d_ws;
  float* x     = ws + OFF_X;
  float* nembB = ws + OFF_NEMB;
  float* casc  = ws + OFF_CASC;
  float* qkv   = ws + OFF_QKV;
  float* sc    = ws + OFF_SC;
  float* ctx   = ws + OFF_CTX;
  float* proj  = ws + OFF_PROJ;
  float* ffb   = ws + OFF_FF;
  float* xout  = (float*)d_out;

  const long long SRow = (long long)cS * 3 * cH;   // per-batch stride inside qkv
  const long long SS   = (long long)cS * cS;

  // 1) token + position embeddings
  embed_kernel<<<cRows, 256, 0, stream>>>(input_ids, tok_emb, pos_emb, x);

  // 2) cascade node embeddings
  {
    const long long total = (long long)cS * cNN * cE;
    cascade_node_kernel<<<(int)((total + 255) / 256), 256, 0, stream>>>(
        ccids, cweights, cmask, node_emb, cw_W, cw_b, nembB);
  }

  // 3) casc = nemb @ fus_W^T + fus_b   (512 x 768 x 3328)
  gemm_bf16_wmma<<<dim3(cH / 64, cS / 64, 1), 128, 0, stream>>>(
      nembB, fus_W, fus_b, casc, cS, cH, cKcas, cKcas, cKcas, cH,
      0, 0, 0, 0, 0, 0, 1, 1.0f, kBias);

  // 4) x = LN(x + casc broadcast over batch), eps 1e-12
  add_ln_kernel<<<cRows, 256, 0, stream>>>(x, casc, cS, ln_g, ln_b, 1e-12f, x);

  // 5) encoder layers
  for (int l = 0; l < cNL; ++l) {
    const float* iW = attn_in_W  + (long long)l * 3 * cH * cH;
    const float* ib = attn_in_b  + (long long)l * 3 * cH;
    const float* oW = attn_out_W + (long long)l * cH * cH;
    const float* ob = attn_out_b + (long long)l * cH;
    const float* f1W = ff1_W + (long long)l * cFF * cH;
    const float* f1b = ff1_b + (long long)l * cFF;
    const float* f2W = ff2_W + (long long)l * cH * cFF;
    const float* f2b = ff2_b + (long long)l * cH;
    const float* g1 = ln1_g + (long long)l * cH;
    const float* b1 = ln1_b + (long long)l * cH;
    const float* g2 = ln2_g + (long long)l * cH;
    const float* b2 = ln2_b + (long long)l * cH;

    // QKV: [2048,768] @ [2304,768]^T + bias
    gemm_bf16_wmma<<<dim3(3 * cH / 64, cRows / 64, 1), 128, 0, stream>>>(
        x, iW, ib, qkv, cRows, 3 * cH, cH, cH, cH, 3 * cH,
        0, 0, 0, 0, 0, 0, 1, 1.0f, kBias);

    // scores[z] = (Q @ K^T) / 8 for z = b*NH + h   (512 x 512 x 64)
    gemm_bf16_wmma<<<dim3(cS / 64, cS / 64, cB * cNH), 128, 0, stream>>>(
        qkv, qkv + cH, nullptr, sc, cS, cS, cHD,
        3 * cH, 3 * cH, cS,
        SRow, cHD, SRow, cHD, (long long)cNH * SS, SS,
        cNH, 0.125f, 0);

    // softmax rows
    softmax_kernel<<<cB * cNH * cS, 256, 0, stream>>>(sc);

    // ctx[z] = att @ V   (512 x 64 x 512), V read transposed
    gemm_bf16_wmma<<<dim3(cHD / 64, cS / 64, cB * cNH), 128, 0, stream>>>(
        sc, qkv + 2 * cH, nullptr, ctx, cS, cHD, cS,
        cS, 3 * cH, cH,
        (long long)cNH * SS, SS, SRow, cHD, (long long)cS * cH, cHD,
        cNH, 1.0f, kTransB);

    // out-projection
    gemm_bf16_wmma<<<dim3(cH / 64, cRows / 64, 1), 128, 0, stream>>>(
        ctx, oW, ob, proj, cRows, cH, cH, cH, cH, cH,
        0, 0, 0, 0, 0, 0, 1, 1.0f, kBias);

    // x = LN(x + attn_out), eps 1e-5
    add_ln_kernel<<<cRows, 256, 0, stream>>>(x, proj, cRows, g1, b1, 1e-5f, x);

    // FFN up + relu
    gemm_bf16_wmma<<<dim3(cFF / 64, cRows / 64, 1), 128, 0, stream>>>(
        x, f1W, f1b, ffb, cRows, cFF, cH, cH, cH, cFF,
        0, 0, 0, 0, 0, 0, 1, 1.0f, kBias | kRelu);

    // FFN down
    gemm_bf16_wmma<<<dim3(cH / 64, cRows / 64, 1), 128, 0, stream>>>(
        ffb, f2W, f2b, proj, cRows, cH, cFF, cFF, cFF, cH,
        0, 0, 0, 0, 0, 0, 1, 1.0f, kBias);

    // x = LN(x + ff_out); final layer writes straight to d_out
    float* dst = (l == cNL - 1) ? xout : x;
    add_ln_kernel<<<cRows, 256, 0, stream>>>(x, proj, cRows, g2, b2, 1e-5f, dst);
  }
}
